// GCN_88450556494350
// MI455X (gfx1250) — matrix operand order repaired and checked by reading, not compiled, not run on hardware
//
#include <hip/hip_runtime.h>

typedef __bf16 v16bf __attribute__((ext_vector_type(16)));
typedef float  v8f   __attribute__((ext_vector_type(8)));

#define NDL 0.5f

// ---------------------------------------------------------------------------
// Utility kernels
// ---------------------------------------------------------------------------
__global__ void k_fill(float* __restrict__ p, float v, long long n) {
  long long i = (long long)blockIdx.x * blockDim.x + threadIdx.x;
  if (i < n) p[i] = v;
}

__global__ void k_count_deg(const int* __restrict__ col, int e, float* __restrict__ deg) {
  int i = blockIdx.x * blockDim.x + threadIdx.x;
  if (i < e) unsafeAtomicAdd(&deg[col[i]], 1.0f);
}

__global__ void k_rsqrt_inplace(float* __restrict__ p, int n) {
  int i = blockIdx.x * blockDim.x + threadIdx.x;
  if (i < n) p[i] = rsqrtf(p[i]);
}

// ---------------------------------------------------------------------------
// Weight pre-pack: f32 [K=128, OUTC] row-major -> bf16 fragment-major layout.
// One thread per (ct, kc, lane); emits that lane's 16 contiguous bf16 values
// so the GEMM can load each B fragment as a single aligned 32-byte vector.
//   n = ct*16 + (lane&15), kbase = kc*32 + (lane>>4)*16, j = 0..15
// ---------------------------------------------------------------------------
template <int OUTC>
__global__ void k_pack_w(const float* __restrict__ W, __bf16* __restrict__ Wp) {
  constexpr int NT = OUTC / 16;
  constexpr int NKC = 4;  // 128/32 k-chunks
  int t = blockIdx.x * blockDim.x + threadIdx.x;
  if (t >= NT * NKC * 32) return;
  int lane = t & 31;
  int kc = (t >> 5) & (NKC - 1);
  int ct = t >> 7;
  int n = ct * 16 + (lane & 15);
  int kbase = kc * 32 + (lane >> 4) * 16;
#pragma unroll
  for (int j = 0; j < 16; ++j)
    Wp[(size_t)t * 16 + j] = (__bf16)W[(size_t)(kbase + j) * OUTC + n];
}

// ---------------------------------------------------------------------------
// Dual-weight WMMA GEMM:  Ya = X@Wa, Yb = X@Wb   (K = 128, bf16, f32 acc)
// Block = 32 rows of X staged once in LDS; 2*OUTC/16 waves; each wave keeps
// its 4 packed B fragments in registers and runs 2 row-tiles (8 WMMAs/wave).
// ---------------------------------------------------------------------------
template <typename TIn, int OUTC>
__global__ __launch_bounds__(OUTC * 4) void k_gemm_dual(
    const TIn* __restrict__ X, const __bf16* __restrict__ Wpa,
    const __bf16* __restrict__ Wpb, float* __restrict__ Ya,
    float* __restrict__ Yb) {
  constexpr int K = 128;
  constexpr int NT = OUTC / 16;       // column tiles per output matrix
  constexpr int NTHREADS = OUTC * 4;  // 2*NT waves * 32
  constexpr int MT = 2;               // row tiles per block
  __shared__ __bf16 As[16 * MT][K + 8];  // pad kills LDS bank conflicts

  const int tid = threadIdx.x;
  const int m0 = blockIdx.x * (16 * MT);

  // Stage the (16*MT) x K A-tile into LDS as bf16 (X read once per block).
  for (int i = tid; i < 16 * MT * K; i += NTHREADS) {
    int r = i >> 7, c = i & 127;
    As[r][c] = (__bf16)(float)X[(size_t)(m0 + r) * K + c];
  }
  __syncthreads();

  const int wave = tid >> 5;
  const int lane = tid & 31;
  const int out_sel = wave / NT;  // 0 -> (Wpa,Ya), 1 -> (Wpb,Yb)
  const int ct = wave % NT;
  const v16bf* __restrict__ Wp =
      reinterpret_cast<const v16bf*>(out_sel ? Wpb : Wpa);
  float* __restrict__ Y = out_sel ? Yb : Ya;

  // Load packed B fragments once; reused for both row tiles.
  v16bf bfrag[K / 32];
#pragma unroll
  for (int kc = 0; kc < K / 32; ++kc)
    bfrag[kc] = Wp[(ct * (K / 32) + kc) * 32 + lane];

  const int mrow = lane & 15;
  const int khiA = (lane >> 4) * 16;  // contiguous lane K offset (16h), matches packed B
  const int n = ct * 16 + (lane & 15);

#pragma unroll
  for (int mt = 0; mt < MT; ++mt) {
    v8f acc = {};
#pragma unroll
    for (int kc = 0; kc < K / 32; ++kc) {
      const int kb = kc * 32;
      v16bf a;
#pragma unroll
      for (int v = 0; v < 8; ++v) {  // K = {0..15} + khiA (contiguous, matches packed B)
        int ka = kb + khiA + v * 2;
        a[2 * v]     = As[mt * 16 + mrow][ka];
        a[2 * v + 1] = As[mt * 16 + mrow][ka + 1];
      }
      acc = __builtin_amdgcn_wmma_f32_16x16x32_bf16(
          false, a, false, bfrag[kc], (short)0, acc, false, false);
    }
    // D layout: VGPR r -> row r + 8*(lane>>4), col = lane&15
    const int rbase = m0 + mt * 16 + (lane >> 4) * 8;
#pragma unroll
    for (int r = 0; r < 8; ++r)
      Y[(size_t)(rbase + r) * OUTC + n] = acc[r];
  }
}

// ---------------------------------------------------------------------------
// Edge scatter: dst[col] += scale * dinv[row]*dinv[col] * src[row]
// ---------------------------------------------------------------------------
template <int LOGC>
__global__ void k_scatter(const float* __restrict__ src,
                          const float* __restrict__ dinv,
                          const int* __restrict__ rows,
                          const int* __restrict__ cols, int nedges, float scale,
                          float* __restrict__ dst) {
  constexpr int C = 1 << LOGC;
  long long idx = (long long)blockIdx.x * blockDim.x + threadIdx.x;
  if (idx >= ((long long)nedges << LOGC)) return;
  int e = (int)(idx >> LOGC);
  int f = (int)(idx & (C - 1));
  int r = rows[e], c = cols[e];
  float w = scale * dinv[r] * dinv[c];
  unsafeAtomicAdd(&dst[((size_t)c << LOGC) + f],
                  w * src[((size_t)r << LOGC) + f]);
}

// self-loop contribution of both branches + biases, fused
template <int LOGC>
__global__ void k_selfloop_bias(const float* __restrict__ srcA,
                                const float* __restrict__ srcB,
                                const float* __restrict__ dinvA,
                                const float* __restrict__ dinvB,
                                const float* __restrict__ bA,
                                const float* __restrict__ bB, int n,
                                float* __restrict__ dst) {
  constexpr int C = 1 << LOGC;
  long long idx = (long long)blockIdx.x * blockDim.x + threadIdx.x;
  if (idx >= ((long long)n << LOGC)) return;
  int i = (int)(idx >> LOGC), f = (int)(idx & (C - 1));
  float wa = dinvA[i]; wa *= wa;
  float wb = dinvB[i]; wb *= wb;
  dst[idx] += wa * srcA[idx] + NDL * wb * srcB[idx] + bA[f] + NDL * bB[f];
}

// ---------------------------------------------------------------------------
// BatchNorm (training stats over node dim), fused ReLU + bf16 output
// ---------------------------------------------------------------------------
__global__ void k_bn_stats(const float* __restrict__ h, int n,
                           float* __restrict__ sum, float* __restrict__ sumsq) {
  __shared__ float s1[256], s2[256];
  int c = threadIdx.x & 127;
  int half = threadIdx.x >> 7;
  float a = 0.f, b = 0.f;
  for (long long r = (long long)blockIdx.x * 2 + half; r < n;
       r += (long long)gridDim.x * 2) {
    float v = h[r * 128 + c];
    a += v;
    b += v * v;
  }
  s1[threadIdx.x] = a;
  s2[threadIdx.x] = b;
  __syncthreads();
  if (half == 0) {
    unsafeAtomicAdd(&sum[c], s1[c] + s1[c + 128]);
    unsafeAtomicAdd(&sumsq[c], s2[c] + s2[c + 128]);
  }
}

__global__ void k_bn_finalize(const float* __restrict__ sum,
                              const float* __restrict__ sumsq,
                              const float* __restrict__ gamma,
                              const float* __restrict__ beta, int n,
                              float* __restrict__ scale,
                              float* __restrict__ shift) {
  int c = threadIdx.x;  // 128 threads
  float inv_n = 1.0f / (float)n;
  float mean = sum[c] * inv_n;
  float var = sumsq[c] * inv_n - mean * mean;
  float sc = gamma[c] * rsqrtf(var + 1e-5f);
  scale[c] = sc;
  shift[c] = beta[c] - mean * sc;
}

__global__ void k_bn_relu_bf16(const float* __restrict__ h,
                               const float* __restrict__ scale,
                               const float* __restrict__ shift, long long total,
                               __bf16* __restrict__ out) {
  long long idx = (long long)blockIdx.x * blockDim.x + threadIdx.x;
  if (idx >= total) return;
  int f = (int)(idx & 127);
  float v = fmaf(h[idx], scale[f], shift[f]);
  out[idx] = (__bf16)fmaxf(v, 0.0f);
}

// ---------------------------------------------------------------------------
// Host orchestration
// ---------------------------------------------------------------------------
static inline int cdiv(long long a, int b) { return (int)((a + b - 1) / b); }

extern "C" void kernel_launch(void* const* d_in, const int* in_sizes, int n_in,
                              void* d_out, int out_size, void* d_ws,
                              size_t ws_size, hipStream_t stream) {
  const float* x    = (const float*)d_in[0];
  const int* adj_l  = (const int*)d_in[1];   // [2,E]: rows then cols
  const int* adj_n  = (const int*)d_in[3];   // adj_nd_low (high arrays unused)
  const float* W1l  = (const float*)d_in[5];
  const float* b1l  = (const float*)d_in[6];
  const float* W1n  = (const float*)d_in[7];
  const float* b1n  = (const float*)d_in[8];
  const float* gam  = (const float*)d_in[9];
  const float* bet  = (const float*)d_in[10];
  const float* W2l  = (const float*)d_in[11];
  const float* b2l  = (const float*)d_in[12];
  const float* W2n  = (const float*)d_in[13];
  const float* b2n  = (const float*)d_in[14];

  const int N = in_sizes[0] / 128;
  const int E = in_sizes[1] / 2;

  // workspace carve-up (256B aligned)
  char* ws = (char*)d_ws;
  size_t off = 0;
  auto take = [&](size_t bytes) {
    size_t r = off;
    off += (bytes + 255) & ~(size_t)255;
    return r;
  };
  __bf16* wp1l = (__bf16*)(ws + take(128 * 128 * 2));
  __bf16* wp1n = (__bf16*)(ws + take(128 * 128 * 2));
  __bf16* wp2l = (__bf16*)(ws + take(128 * 64 * 2));
  __bf16* wp2n = (__bf16*)(ws + take(128 * 64 * 2));
  float* dinv_l = (float*)(ws + take((size_t)N * 4));
  float* dinv_n = (float*)(ws + take((size_t)N * 4));
  float* stats  = (float*)(ws + take(512 * 4));  // sum|sumsq|scale|shift
  float* XWl = (float*)(ws + take((size_t)N * 128 * 4));
  float* XWn = (float*)(ws + take((size_t)N * 128 * 4));
  float* h   = (float*)(ws + take((size_t)N * 128 * 4));
  // reuse: after layer-1 scatter, XWl holds bf16 h; XWn holds layer-2 GEMMs
  __bf16* hb  = (__bf16*)XWl;
  float* HWl  = XWn;
  float* HWn  = XWn + (size_t)N * 64;
  float* out  = (float*)d_out;

  // 1) weights -> packed bf16 fragment layout
  k_pack_w<128><<<4, 256, 0, stream>>>(W1l, wp1l);
  k_pack_w<128><<<4, 256, 0, stream>>>(W1n, wp1n);
  k_pack_w<64><<<2, 256, 0, stream>>>(W2l, wp2l);
  k_pack_w<64><<<2, 256, 0, stream>>>(W2n, wp2n);

  // 2) init: deg=1 (self loops), h=0, stats=0, out=0
  k_fill<<<cdiv(N, 256), 256, 0, stream>>>(dinv_l, 1.0f, N);
  k_fill<<<cdiv(N, 256), 256, 0, stream>>>(dinv_n, 1.0f, N);
  k_fill<<<cdiv((long long)N * 128, 256), 256, 0, stream>>>(h, 0.0f,
                                                            (long long)N * 128);
  k_fill<<<1, 256, 0, stream>>>(stats, 0.0f, 256);
  k_fill<<<cdiv((long long)N * 64, 256), 256, 0, stream>>>(out, 0.0f,
                                                           (long long)N * 64);

  // 3) degrees -> dinv
  k_count_deg<<<cdiv(E, 256), 256, 0, stream>>>(adj_l + E, E, dinv_l);
  k_count_deg<<<cdiv(E, 256), 256, 0, stream>>>(adj_n + E, E, dinv_n);
  k_rsqrt_inplace<<<cdiv(N, 256), 256, 0, stream>>>(dinv_l, N);
  k_rsqrt_inplace<<<cdiv(N, 256), 256, 0, stream>>>(dinv_n, N);

  // 4) layer-1 dual GEMM (bf16 WMMA): XWl = x@W1_low, XWn = x@W1_nd
  k_gemm_dual<float, 128><<<N / 32, 512, 0, stream>>>(x, wp1l, wp1n, XWl, XWn);

  // 5) propagate both branches into h, then self loops + biases
  k_scatter<7><<<cdiv((long long)E * 128, 256), 256, 0, stream>>>(
      XWl, dinv_l, adj_l, adj_l + E, E, 1.0f, h);
  k_scatter<7><<<cdiv((long long)E * 128, 256), 256, 0, stream>>>(
      XWn, dinv_n, adj_n, adj_n + E, E, NDL, h);
  k_selfloop_bias<7><<<cdiv((long long)N * 128, 256), 256, 0, stream>>>(
      XWl, XWn, dinv_l, dinv_n, b1l, b1n, N, h);

  // 6) BatchNorm + ReLU -> bf16 activations
  k_bn_stats<<<256, 256, 0, stream>>>(h, N, stats, stats + 128);
  k_bn_finalize<<<1, 128, 0, stream>>>(stats, stats + 128, gam, bet, N,
                                       stats + 256, stats + 384);
  k_bn_relu_bf16<<<cdiv((long long)N * 128, 256), 256, 0, stream>>>(
      h, stats + 256, stats + 384, (long long)N * 128, hb);

  // 7) layer-2 dual GEMM (bf16 WMMA): HWl = h@W2_low, HWn = h@W2_nd
  k_gemm_dual<__bf16, 64><<<N / 32, 256, 0, stream>>>(hb, wp2l, wp2n, HWl, HWn);

  // 8) propagate into output + self loops + biases
  k_scatter<6><<<cdiv((long long)E * 64, 256), 256, 0, stream>>>(
      HWl, dinv_l, adj_l, adj_l + E, E, 1.0f, out);
  k_scatter<6><<<cdiv((long long)E * 64, 256), 256, 0, stream>>>(
      HWn, dinv_n, adj_n, adj_n + E, E, NDL, out);
  k_selfloop_bias<6><<<cdiv((long long)N * 64, 256), 256, 0, stream>>>(
      HWl, HWn, dinv_l, dinv_n, b2l, b2n, N, out);
}
